// NeighborCompressedNN_79577154060338
// MI455X (gfx1250) — matrix operand order repaired and checked by reading, not compiled, not run on hardware
//
#include <hip/hip_runtime.h>
#include <hip/hip_bf16.h>
#include <math.h>

// ---------------------------------------------------------------------------
// NeighborCompressedNN for MI455X (gfx1250, wave32, WMMA)
//   B=1024 queries, N=200000 data rows, F=64, C=16, H=128, K=32 neighbors
//   Ranking key: ||Xd[n]||^2 - 2*x.Xd[n]   (||x||^2 is constant per query row)
//   GEMM in bf16 via v_wmma_f32_16x16x32_bf16 (4 per loop trip), accumulator
//   seeded with norms, -2 folded into A. Ballot-guarded streaming top-32.
//   Neighbor gather uses the gfx1250 async global->LDS path (ASYNCcnt).
// ---------------------------------------------------------------------------

typedef __attribute__((ext_vector_type(16))) __bf16 v16bf;
typedef __attribute__((ext_vector_type(8)))  __bf16 v8bf;
typedef __attribute__((ext_vector_type(8)))  float  v8f;

#define N_DATA   200000
#define B_ROWS   1024
#define F_DIM    64
#define C_DIM    16
#define H_DIM    128
#define K_NB     32
#define N_TILES  (N_DATA / 16)
#define N_PAIRS  (N_TILES / 2)
#define INF_F    3.0e38f

// ---------------------------------------------------------------------------
// Kernel 0: X_data fp32 -> bf16 copy + per-row squared norms.
// ---------------------------------------------------------------------------
__launch_bounds__(256)
__global__ void prep_kernel(const float* __restrict__ X,
                            __bf16* __restrict__ Xb,
                            float* __restrict__ norms) {
    const int row = blockIdx.x * 16 + (threadIdx.x >> 4);
    const int j   = threadIdx.x & 15;
    const float4 q = ((const float4*)(X + (size_t)row * F_DIM))[j];
    __bf16* dst = Xb + (size_t)row * F_DIM + j * 4;
    dst[0] = (__bf16)q.x; dst[1] = (__bf16)q.y;
    dst[2] = (__bf16)q.z; dst[3] = (__bf16)q.w;
    float p = q.x * q.x + q.y * q.y + q.z * q.z + q.w * q.w;
    #pragma unroll
    for (int off = 8; off; off >>= 1) p += __shfl_xor(p, off, 32);
    if (j == 0) norms[row] = p;
}

// ---------------------------------------------------------------------------
// Rare-path reservoir insert for one 16x16 D tile.
// D layout: VGPR v -> row v (lanes 0-15) / v+8 (lanes 16-31), col = lane&15.
// ---------------------------------------------------------------------------
__device__ __forceinline__ void topk_insert_tile(
        const v8f& c, float thr[8], const int n0, const int lane,
        float (* __restrict__ rval)[K_NB], int (* __restrict__ ridx)[K_NB]) {
    #pragma unroll
    for (int v = 0; v < 8; ++v) {
        const float r = c[v];
        unsigned mask = __builtin_amdgcn_ballot_w32(r < thr[v]);
        while (mask) {                      // ~k*ln(N/k) inserts per row total
            const int src = __ffs(mask) - 1;
            mask &= mask - 1;
            const float val = __shfl(r, src, 32);
            const int m   = v + ((src >> 4) << 3);
            const int col = n0 + (src & 15);
            const float bv = rval[m][lane];
            float mv = bv; int mi = lane;
            #pragma unroll
            for (int off = 16; off; off >>= 1) {
                const float ov = __shfl_xor(mv, off, 32);
                const int   oi = __shfl_xor(mi, off, 32);
                if (ov > mv || (ov == mv && oi > mi)) { mv = ov; mi = oi; }
            }
            if (val < mv) {
                float nv = bv;
                if (lane == mi) {
                    rval[m][lane] = val;
                    ridx[m][lane] = col;
                    nv = val;
                }
                #pragma unroll
                for (int off = 16; off; off >>= 1)
                    nv = fmaxf(nv, __shfl_xor(nv, off, 32));
                if ((lane >> 4) == (m >> 3)) thr[v] = nv;
            }
        }
    }
}

// ---------------------------------------------------------------------------
// Kernel 1: fused distance GEMM (bf16 WMMA) + streaming top-32 per query row.
// 64 blocks x 256 threads (8 wave32). Block owns 16 query rows; waves stride
// over pairs of 16-wide N tiles (4 WMMAs / trip, one fused ballot check).
// ---------------------------------------------------------------------------
__launch_bounds__(256)
__global__ void dist_topk_kernel(const float* __restrict__ x,
                                 const __bf16* __restrict__ Xb,
                                 const float* __restrict__ norms,
                                 int* __restrict__ topk) {
    __shared__ __align__(16) __bf16 xs[16][F_DIM];     // -2*x tile, bf16
    __shared__ float cval[8][16][K_NB];                // per-wave reservoirs
    __shared__ int   cidx[8][16][K_NB];

    const int tid  = threadIdx.x;
    const int wid  = tid >> 5;
    const int lane = tid & 31;
    const int half = lane >> 4;
    const int lm   = lane & 15;
    const int row0 = blockIdx.x * 16;

    for (int i = tid; i < 16 * F_DIM; i += 256) {
        const int r = i >> 6, f = i & 63;
        xs[r][f] = (__bf16)(-2.0f * x[(size_t)(row0 + r) * F_DIM + f]);
    }
    for (int i = tid; i < 8 * 16 * K_NB; i += 256) {
        (&cval[0][0][0])[i] = INF_F;
        (&cidx[0][0][0])[i] = 0;
    }
    __syncthreads();

    // A fragments: lane holds 16 contiguous K (k = 16*half + s), matching B.
    const int k0 = half * 16;
    const v8bf lo0 = *(const v8bf*)&xs[lm][k0];
    const v8bf hi0 = *(const v8bf*)&xs[lm][k0 + 8];
    const v8bf lo1 = *(const v8bf*)&xs[lm][32 + k0];
    const v8bf hi1 = *(const v8bf*)&xs[lm][32 + k0 + 8];
    const v16bf a0 = __builtin_shufflevector(lo0, hi0, 0,1,2,3,4,5,6,7,8,9,10,11,12,13,14,15);
    const v16bf a1 = __builtin_shufflevector(lo1, hi1, 0,1,2,3,4,5,6,7,8,9,10,11,12,13,14,15);

    float thr[8];
    #pragma unroll
    for (int v = 0; v < 8; ++v) thr[v] = INF_F;

    const int koff = half * 16;   // B 32x16 bf16: lane holds 16 contiguous K

    float (* __restrict__ rval)[K_NB] = cval[wid];
    int   (* __restrict__ ridx)[K_NB] = cidx[wid];

    // 6250 tile-pairs strided across 8 waves (waves 0,1 absorb the remainder)
    for (int p = wid; p < N_PAIRS; p += 8) {
        const int n0 = p * 32;            // tile 0: cols n0..n0+15
        const int n1 = n0 + 16;           // tile 1: cols n1..n1+15

        const __bf16* br0 = Xb + (size_t)(n0 + lm) * F_DIM;
        const __bf16* br1 = Xb + (size_t)(n1 + lm) * F_DIM;
        const v16bf b00 = *(const v16bf*)(br0 + koff);
        const v16bf b01 = *(const v16bf*)(br0 + 32 + koff);
        const v16bf b10 = *(const v16bf*)(br1 + koff);
        const v16bf b11 = *(const v16bf*)(br1 + 32 + koff);
        const float nr0 = norms[n0 + lm];
        const float nr1 = norms[n1 + lm];
        if (p + 8 < N_PAIRS)   // prefetch this wave's next pair (global_prefetch_b8)
            __builtin_prefetch(Xb + (size_t)(n0 + 256 + lm) * F_DIM, 0, 1);

        v8f c0, c1;
        #pragma unroll
        for (int v = 0; v < 8; ++v) { c0[v] = nr0; c1[v] = nr1; }
        c0 = __builtin_amdgcn_wmma_f32_16x16x32_bf16(false, a0, false, b00,
                                                     (short)0, c0, false, false);
        c1 = __builtin_amdgcn_wmma_f32_16x16x32_bf16(false, a0, false, b10,
                                                     (short)0, c1, false, false);
        c0 = __builtin_amdgcn_wmma_f32_16x16x32_bf16(false, a1, false, b01,
                                                     (short)0, c0, false, false);
        c1 = __builtin_amdgcn_wmma_f32_16x16x32_bf16(false, a1, false, b11,
                                                     (short)0, c1, false, false);

        // Fused fast-path check: one ballot/branch over all 16 result VGPRs.
        bool pred = false;
        #pragma unroll
        for (int v = 0; v < 8; ++v) pred = pred | (c0[v] < thr[v]) | (c1[v] < thr[v]);
        if (__builtin_amdgcn_ballot_w32(pred)) {
            topk_insert_tile(c0, thr, n0, lane, rval, ridx);
            topk_insert_tile(c1, thr, n1, lane, rval, ridx);
        }
    }
    __syncthreads();

    // Merge 8 per-wave reservoirs (256 candidates) -> final 32 per row.
    // Output order is irrelevant: downstream gather+sum is permutation invariant.
    for (int mm = 0; mm < 2; ++mm) {
        const int m = wid * 2 + mm;
        for (int it = 0; it < K_NB; ++it) {
            float best = cval[0][m][lane]; int bw = 0;
            #pragma unroll
            for (int w = 1; w < 8; ++w) {
                const float v2 = cval[w][m][lane];
                if (v2 < best) { best = v2; bw = w; }
            }
            float rv = best; int rl = lane;
            #pragma unroll
            for (int off = 16; off; off >>= 1) {
                const float ov = __shfl_xor(rv, off, 32);
                const int   ol = __shfl_xor(rl, off, 32);
                if (ov < rv || (ov == rv && ol < rl)) { rv = ov; rl = ol; }
            }
            if (lane == rl) {
                cval[bw][m][lane] = INF_F;
                topk[(size_t)(row0 + m) * K_NB + it] = cidx[bw][m][lane];
            }
        }
    }
}

// ---------------------------------------------------------------------------
// Kernel 2: gather neighbors (+y), gated tanh, sum over K -> agg[B,16]
// Feature gather goes through the gfx1250 async global->LDS path: per-lane
// global address -> per-lane LDS address, tracked by ASYNCcnt (no VGPR bounce).
// nf row stride = 68 floats (272B) so every 16B async chunk is 16B-aligned.
// ---------------------------------------------------------------------------
__launch_bounds__(128)
__global__ void gate_kernel(const float* __restrict__ X,
                            const float* __restrict__ y,
                            const float* __restrict__ Wg,
                            const float* __restrict__ bg,
                            const int* __restrict__ topk,
                            float* __restrict__ agg) {
    __shared__ __align__(16) float nf[K_NB][68];   // [0..63]=features, [64]=y
    __shared__ int ids[K_NB];
    __shared__ float part[8][C_DIM];
    const int t = threadIdx.x;
    const int b = blockIdx.x;
    if (t < K_NB) ids[t] = topk[(size_t)b * K_NB + t];
    __syncthreads();

    // 32 rows x 16 chunks of 16B = 512 async units; 128 threads -> 4 rounds.
    #pragma unroll
    for (int u = t; u < K_NB * 16; u += 128) {
        const int r  = u >> 4;
        const int ch = u & 15;
        const unsigned lds_addr =
            (unsigned)(uintptr_t)&nf[r][0] + (unsigned)(ch * 16);
        const float* gaddr = X + (size_t)ids[r] * F_DIM + ch * 4;
        asm volatile("global_load_async_to_lds_b128 %0, %1, off"
                     :: "v"(lds_addr), "v"(gaddr) : "memory");
    }
    if (t < K_NB) nf[t][F_DIM] = y[ids[t]];        // y column (regular load)
    asm volatile("s_wait_asynccnt 0x0" ::: "memory");
    __syncthreads();

    const int c = t & 15;
    const int g = t >> 4;
    float acc = 0.f;
    for (int kk = g; kk < K_NB; kk += 8) {
        float s = bg[c];
        #pragma unroll 4
        for (int f = 0; f < F_DIM + 1; ++f) s += nf[kk][f] * Wg[f * C_DIM + c];
        acc += tanhf(s);
    }
    part[g][c] = acc;
    __syncthreads();
    if (t < C_DIM) {
        float s = 0.f;
        #pragma unroll
        for (int g2 = 0; g2 < 8; ++g2) s += part[g2][t];
        agg[(size_t)b * C_DIM + t] = s;
    }
}

// ---------------------------------------------------------------------------
// Kernel 3: o = sigmoid(tanh([x,agg] @ W1 + b1) @ W_last + b_last)
// ---------------------------------------------------------------------------
__launch_bounds__(128)
__global__ void mlp_kernel(const float* __restrict__ x,
                           const float* __restrict__ agg,
                           const float* __restrict__ W1,
                           const float* __restrict__ b1,
                           const float* __restrict__ Wl,
                           const float* __restrict__ bl,
                           float* __restrict__ out) {
    __shared__ float xo[F_DIM + C_DIM];
    __shared__ float partial[4];
    const int b = blockIdx.x, t = threadIdx.x;
    if (t < F_DIM)              xo[t] = x[(size_t)b * F_DIM + t];
    else if (t < F_DIM + C_DIM) xo[t] = agg[(size_t)b * C_DIM + (t - F_DIM)];
    __syncthreads();
    float s = b1[t];
    #pragma unroll 4
    for (int f = 0; f < F_DIM + C_DIM; ++f) s += xo[f] * W1[f * H_DIM + t];
    float o = tanhf(s) * Wl[t];
    #pragma unroll
    for (int off = 16; off; off >>= 1) o += __shfl_xor(o, off, 32);
    if ((t & 31) == 0) partial[t >> 5] = o;
    __syncthreads();
    if (t == 0) {
        const float z = partial[0] + partial[1] + partial[2] + partial[3] + bl[0];
        out[b] = 1.f / (1.f + expf(-z));
    }
}

// ---------------------------------------------------------------------------
extern "C" void kernel_launch(void* const* d_in, const int* in_sizes, int n_in,
                              void* d_out, int out_size, void* d_ws, size_t ws_size,
                              hipStream_t stream) {
    const float* x      = (const float*)d_in[0];
    const float* X_data = (const float*)d_in[1];
    const float* y      = (const float*)d_in[2];
    const float* W_gate = (const float*)d_in[3];
    const float* b_gate = (const float*)d_in[4];
    const float* W1     = (const float*)d_in[5];
    const float* b1     = (const float*)d_in[6];
    const float* W_last = (const float*)d_in[7];
    const float* b_last = (const float*)d_in[8];
    float* out = (float*)d_out;

    char* ws = (char*)d_ws;
    __bf16* Xb   = (__bf16*)ws;                                   // 25,600,000 B
    float*  nrm  = (float*)(ws + 25600000);                       //    800,000 B
    int*    topk = (int*)  (ws + 25600000 + 800000);              //    131,072 B
    float*  agg  = (float*)(ws + 25600000 + 800000 + 131072);     //     65,536 B

    prep_kernel<<<N_DATA / 16, 256, 0, stream>>>(X_data, Xb, nrm);
    dist_topk_kernel<<<B_ROWS / 16, 256, 0, stream>>>(x, Xb, nrm, topk);
    gate_kernel<<<B_ROWS, 128, 0, stream>>>(X_data, y, W_gate, b_gate, topk, agg);
    mlp_kernel<<<B_ROWS, 128, 0, stream>>>(x, agg, W1, b1, W_last, b_last, out);
}
